// BatchGATLayer_27977416966564
// MI455X (gfx1250) — matrix operand order repaired and checked by reading, not compiled, not run on hardware
//
#include <hip/hip_runtime.h>
#include <hip/hip_bf16.h>

#define NNODES   20000
#define INDIM    128
#define DMODEL   64
#define NHEADS   4
#define TSTEPS   8
#define NEDGES   320000
#define ETOT     (NEDGES + NNODES)     // self loops appended
#define HC       (NHEADS * DMODEL)     // 256
#define NEGSLOPE 0.2f
#define KSTEPS   (INDIM / 32)          // 4
#define CTILES   (HC / 16)             // 16

typedef __attribute__((ext_vector_type(16))) __bf16 v16bf;
typedef __attribute__((ext_vector_type(8)))  float  v8f;

// ---- monotonic float<->uint mapping for atomicMax-based segment max ----
__device__ __forceinline__ unsigned f2ord(float f) {
    unsigned u = __float_as_uint(f);
    return (u & 0x80000000u) ? ~u : (u | 0x80000000u);
}
__device__ __forceinline__ float ord2f(unsigned u) {
    unsigned b = (u & 0x80000000u) ? (u & 0x7FFFFFFFu) : ~u;
    return __uint_as_float(b);
}

// ---- pre-pack W into per-lane B fragments: Bpack[(colTile*KSTEPS+kstep)*32+lane][16 bf16]
// B frag layout (ISA 7.12.2): col = lane&15, lanes 0-15 K=k0..k0+7,k0+16..k0+23, lanes 16-31 K=k0+8..k0+15,k0+24..k0+31
__global__ void pack_b(const float* __restrict__ W, __bf16* __restrict__ Bpack) {
    int idx = blockIdx.x * blockDim.x + threadIdx.x;
    if (idx >= CTILES * KSTEPS * 32) return;
    int lane = idx & 31, kstep = (idx >> 5) & (KSTEPS - 1), colTile = idx >> 7;
    int col = colTile * 16 + (lane & 15);
    int kb  = kstep * 32 + ((lane & 16) ? 8 : 0);
    __bf16* dst = Bpack + (size_t)idx * 16;
    #pragma unroll
    for (int i = 0; i < 16; ++i)
        dst[i] = (__bf16)W[(size_t)(kb + i + (i >= 8 ? 8 : 0)) * HC + col];
}

// ---- out[n][t][c] = bias[c] (accumulation base) ----
__global__ void init_out(float* __restrict__ out, const float* __restrict__ bias) {
    int i = blockIdx.x * blockDim.x + threadIdx.x;
    if (i < NNODES * TSTEPS * DMODEL) out[i] = bias[i & (DMODEL - 1)];
}

// ---- per-step reset of segment-max / denom ----
__global__ void init_nh(unsigned* __restrict__ amaxU, float* __restrict__ denom) {
    int i = blockIdx.x * blockDim.x + threadIdx.x;
    if (i < NNODES * NHEADS) { amaxU[i] = 0u; denom[i] = 0.f; }
}

// ---- bf16 WMMA GEMM: h_perm[n][c*4+h] = (x_t @ W)[n][h*64+c] ----
// block = 4 waves, covers 16 rows x 256 cols; each wave: 4 col tiles, A frag reused 4x
__global__ void gemm_hperm(const float* __restrict__ x, const v16bf* __restrict__ Bpack,
                           float* __restrict__ hperm, int t) {
    const int lane    = threadIdx.x & 31;
    const int wave    = threadIdx.x >> 5;            // 0..3
    const int rowTile = blockIdx.x;                  // 0..1249
    const int hi      = (lane & 16) ? 1 : 0;
    const int n       = rowTile * 16 + (lane & 15);
    const float* xrow = x + ((size_t)n * TSTEPS + t) * INDIM;

    v8f acc0 = {}, acc1 = {}, acc2 = {}, acc3 = {};
    const int ct0 = wave * 4;                        // this wave's first col tile

    #pragma unroll
    for (int ks = 0; ks < KSTEPS; ++ks) {
        // A frag (16-bit A 16x32, ISA 7.12.2): lane<16 -> K 0..7,16..23 ; lane>=16 -> K 8..15,24..31
        v16bf a;
        const int klo = ks * 32 + (hi ? 8 : 0);
        #pragma unroll
        for (int j = 0; j < 8; ++j) {
            int kp = (j < 4) ? (klo + 2 * j) : (klo + 16 + 2 * (j - 4));
            a[2 * j]     = (__bf16)xrow[kp];
            a[2 * j + 1] = (__bf16)xrow[kp + 1];
        }
        // 4 pre-packed B frags, each one 32B contiguous load
        v16bf b0 = Bpack[((ct0 + 0) * KSTEPS + ks) * 32 + lane];
        v16bf b1 = Bpack[((ct0 + 1) * KSTEPS + ks) * 32 + lane];
        v16bf b2 = Bpack[((ct0 + 2) * KSTEPS + ks) * 32 + lane];
        v16bf b3 = Bpack[((ct0 + 3) * KSTEPS + ks) * 32 + lane];
        acc0 = __builtin_amdgcn_wmma_f32_16x16x32_bf16(false, a, false, b0, (short)0, acc0, false, false);
        acc1 = __builtin_amdgcn_wmma_f32_16x16x32_bf16(false, a, false, b1, (short)0, acc1, false, false);
        acc2 = __builtin_amdgcn_wmma_f32_16x16x32_bf16(false, a, false, b2, (short)0, acc2, false, false);
        acc3 = __builtin_amdgcn_wmma_f32_16x16x32_bf16(false, a, false, b3, (short)0, acc3, false, false);
    }

    // C/D layout: VGPR r -> M = r + (lane>=16 ? 8 : 0), N = lane&15
    // store head-interleaved: global col j = h*64+c  ->  offset c*4+h
    #pragma unroll
    for (int cc = 0; cc < 4; ++cc) {
        const v8f acc = (cc == 0) ? acc0 : (cc == 1) ? acc1 : (cc == 2) ? acc2 : acc3;
        const int col = (ct0 + cc) * 16 + (lane & 15);
        const int c = col & 63, hh = col >> 6;
        #pragma unroll
        for (int r = 0; r < 8; ++r) {
            int nn = rowTile * 16 + r + (hi ? 8 : 0);
            hperm[(size_t)nn * HC + c * 4 + hh] = acc[r];
        }
    }
}

// ---- a_src[n][h], a_dst[n][h] from head-interleaved h ----
__global__ void attn_coef(const float* __restrict__ hperm, const float* __restrict__ att_src,
                          const float* __restrict__ att_dst, float* __restrict__ a_src,
                          float* __restrict__ a_dst) {
    int idx = blockIdx.x * blockDim.x + threadIdx.x;
    if (idx >= NNODES * NHEADS) return;
    int n = idx >> 2, h = idx & 3;
    const float* hp = hperm + (size_t)n * HC + h;
    float ss = 0.f, sd = 0.f;
    #pragma unroll 8
    for (int c = 0; c < DMODEL; ++c) {
        float v = hp[c * 4];
        ss += v * att_src[h * DMODEL + c];
        sd += v * att_dst[h * DMODEL + c];
    }
    a_src[idx] = ss;
    a_dst[idx] = sd;
}

// ---- pass 1: alpha = leakyrelu(a_src[src]+a_dst[dst]); segment max via atomicMax ----
__global__ void edge_alpha_max(const int* __restrict__ ei, const float* __restrict__ a_src,
                               const float* __restrict__ a_dst, float* __restrict__ alphaBuf,
                               unsigned* __restrict__ amaxU) {
    int idx = blockIdx.x * blockDim.x + threadIdx.x;
    if (idx >= ETOT * NHEADS) return;
    int e = idx >> 2, h = idx & 3;
    int s, d;
    if (e < NEDGES) { s = ei[e]; d = ei[NEDGES + e]; }
    else            { s = d = e - NEDGES; }
    float al = a_src[s * 4 + h] + a_dst[d * 4 + h];
    al = (al > 0.f) ? al : NEGSLOPE * al;
    alphaBuf[idx] = al;
    atomicMax(&amaxU[d * 4 + h], f2ord(al));
}

// ---- pass 2: ex = exp(alpha - amax[dst]); denom += ex ----
__global__ void edge_exp_sum(const int* __restrict__ ei, float* __restrict__ alphaBuf,
                             const unsigned* __restrict__ amaxU, float* __restrict__ denom) {
    int idx = blockIdx.x * blockDim.x + threadIdx.x;
    if (idx >= ETOT * NHEADS) return;
    int e = idx >> 2, h = idx & 3;
    int d = (e < NEDGES) ? ei[NEDGES + e] : (e - NEDGES);
    float ex = __expf(alphaBuf[idx] - ord2f(amaxU[d * 4 + h]));
    alphaBuf[idx] = ex;                 // reuse buffer as ex
    atomicAdd(&denom[d * 4 + h], ex);
}

// ---- pass 3: one wave per edge; head-mean folded into per-channel atomicAdd ----
__global__ void edge_aggregate(const int* __restrict__ ei, const float4* __restrict__ h4,
                               const float4* __restrict__ ex4, const float4* __restrict__ denom4,
                               float* __restrict__ out, int t) {
    int e = (blockIdx.x * blockDim.x + threadIdx.x) >> 5;
    if (e >= ETOT) return;
    int lane = threadIdx.x & 31;
    int s, d;
    if (e < NEDGES) { s = ei[e]; d = ei[NEDGES + e]; }
    else            { s = d = e - NEDGES; }

    float4 exv = ex4[e];
    float4 dnv = denom4[d];
    const float inv = 0.25f;            // mean over 4 heads
    float a0 = exv.x / (dnv.x + 1e-16f) * inv;
    float a1 = exv.y / (dnv.y + 1e-16f) * inv;
    float a2 = exv.z / (dnv.z + 1e-16f) * inv;
    float a3 = exv.w / (dnv.w + 1e-16f) * inv;

    const float4* hrow = h4 + (size_t)s * DMODEL;   // h_perm[s][c] = 4 heads packed
    float* orow = out + ((size_t)d * TSTEPS + t) * DMODEL;

    float4 hv = hrow[lane];
    atomicAdd(&orow[lane],      hv.x * a0 + hv.y * a1 + hv.z * a2 + hv.w * a3);
    hv = hrow[lane + 32];
    atomicAdd(&orow[lane + 32], hv.x * a0 + hv.y * a1 + hv.z * a2 + hv.w * a3);
}

extern "C" void kernel_launch(void* const* d_in, const int* in_sizes, int n_in,
                              void* d_out, int out_size, void* d_ws, size_t ws_size,
                              hipStream_t stream) {
    const float* x       = (const float*)d_in[0];
    const int*   ei      = (const int*)  d_in[1];
    const float* W       = (const float*)d_in[2];
    const float* att_src = (const float*)d_in[3];
    const float* att_dst = (const float*)d_in[4];
    const float* bias    = (const float*)d_in[5];
    float* out = (float*)d_out;

    char* wsb = (char*)d_ws;
    size_t off = 0;
    auto carve = [&](size_t bytes) { void* p = wsb + off; off += (bytes + 255) & ~(size_t)255; return p; };

    float*    hperm  = (float*)   carve((size_t)NNODES * HC * 4);
    __bf16*   Bpack  = (__bf16*)  carve((size_t)CTILES * KSTEPS * 32 * 16 * 2);
    float*    a_src  = (float*)   carve((size_t)NNODES * NHEADS * 4);
    float*    a_dst  = (float*)   carve((size_t)NNODES * NHEADS * 4);
    unsigned* amaxU  = (unsigned*)carve((size_t)NNODES * NHEADS * 4);
    float*    denom  = (float*)   carve((size_t)NNODES * NHEADS * 4);
    float*    exbuf  = (float*)   carve((size_t)ETOT * NHEADS * 4);

    pack_b<<<(CTILES * KSTEPS * 32 + 255) / 256, 256, 0, stream>>>(W, Bpack);
    init_out<<<(NNODES * TSTEPS * DMODEL + 255) / 256, 256, 0, stream>>>(out, bias);

    const int nhBlocks  = (NNODES * NHEADS + 255) / 256;
    const int ehBlocks  = (ETOT * NHEADS + 255) / 256;
    const int aggBlocks = (ETOT * 32 + 255) / 256;

    for (int t = 0; t < TSTEPS; ++t) {
        gemm_hperm<<<NNODES / 16, 128, 0, stream>>>(x, (const v16bf*)Bpack, hperm, t);
        attn_coef<<<nhBlocks, 256, 0, stream>>>(hperm, att_src, att_dst, a_src, a_dst);
        init_nh<<<nhBlocks, 256, 0, stream>>>(amaxU, denom);
        edge_alpha_max<<<ehBlocks, 256, 0, stream>>>(ei, a_src, a_dst, exbuf, amaxU);
        edge_exp_sum<<<ehBlocks, 256, 0, stream>>>(ei, exbuf, amaxU, denom);
        edge_aggregate<<<aggBlocks, 256, 0, stream>>>(ei, (const float4*)hperm,
                                                      (const float4*)exbuf,
                                                      (const float4*)denom, out, t);
    }
}
